// Surge_9663676416624
// MI455X (gfx1250) — matrix operand order repaired and checked by reading, not compiled, not run on hardware
//
#include <hip/hip_runtime.h>
#include <hip/hip_bf16.h>

// ---------------- model dims ----------------
#define B_   256
#define L_   200
#define D_   128
#define H_   128
#define BN_EPS 1e-4f

typedef __attribute__((ext_vector_type(16))) __bf16   v16bf;
typedef __attribute__((ext_vector_type(8)))  float    v8f;
typedef __attribute__((ext_vector_type(4)))  unsigned v4u;
typedef __attribute__((ext_vector_type(8)))  unsigned v8u;

__device__ __forceinline__ unsigned pk_bf16(float a, float b) {
    unsigned ua = __builtin_bit_cast(unsigned, a);
    ua += 0x7FFFu + ((ua >> 16) & 1u);              // RNE
    unsigned ub = __builtin_bit_cast(unsigned, b);
    ub += 0x7FFFu + ((ub >> 16) & 1u);
    return (ua >> 16) | (ub & 0xFFFF0000u);
}

// two 16B loads (K-contiguous bf16 pairs) -> one WMMA fragment
__device__ __forceinline__ v16bf ld_frag_lds(const unsigned* rowp, int o0, int o1) {
    v4u lo = *(const v4u*)(rowp + o0);
    v4u hi = *(const v4u*)(rowp + o1);
    v8u t;
    t[0] = lo[0]; t[1] = lo[1]; t[2] = lo[2]; t[3] = lo[3];
    t[4] = hi[0]; t[5] = hi[1]; t[6] = hi[2]; t[7] = hi[3];
    return __builtin_bit_cast(v16bf, t);
}
__device__ __forceinline__ v16bf ld_frag_glb(const unsigned* rowp, int o0, int o1) {
    v4u lo = *(const v4u*)(rowp + o0);
    v4u hi = *(const v4u*)(rowp + o1);
    v8u t;
    t[0] = lo[0]; t[1] = lo[1]; t[2] = lo[2]; t[3] = lo[3];
    t[4] = hi[0]; t[5] = hi[1]; t[6] = hi[2]; t[7] = hi[3];
    return __builtin_bit_cast(v16bf, t);
}

// =====================================================================
// bf16-WMMA GEMM, templated: C[b] = op(A[b]) @ op(B[b]) (+epilogue)
//   block = 128 threads = 4 wave32, WG tile 32x32, K staged 32 at a time.
//   LDS tiles are packed bf16: A as [m][k], B transposed to [n][k], so
//   every fragment is two ds_load_b128.
//   AMODE 1 : A(row r, col k; K=512) built on the fly: [ai, q, ai-q, ai*q]
//   BTRANS  : B given as (N,K) row-major (S = Xf Xf^T)
//   EPI 1   : += bias[n];  EPI 2: leaky(acc + Cadd[m,n])
// =====================================================================
template <int AMODE, int BTRANS, int EPI>
__global__ void __launch_bounds__(128)
k_gemm_t(const float* __restrict__ A, const float* __restrict__ Bm, float* __restrict__ C,
         int M, int N, int K,
         int lda, int ldb, int ldc,
         long sAb, long sBb, long sCb,
         const float* __restrict__ ai, const float* __restrict__ qv, int q2d, int Lq,
         const float* __restrict__ bias, const float* __restrict__ Cadd, long sDb)
{
    __shared__ __align__(16) unsigned As32[32][20];   // bf16 pairs, [m][kpair]
    __shared__ __align__(16) unsigned Bs32[32][20];   // bf16 pairs, [n][kpair]
    const int b    = blockIdx.z;
    const int M0   = blockIdx.x * 32;
    const int N0   = blockIdx.y * 32;
    const int tid  = threadIdx.x;
    const int lane = tid & 31;
    const int w    = tid >> 5;
    const int wm   = (w >> 1) << 4;
    const int wn   = (w & 1) << 4;
    const int g    = lane >> 4;
    const int l15  = lane & 15;

    auto fetchA = [&](int m, int k) -> float {
        if (m >= M || k >= K) return 0.f;
        if (AMODE == 0) return A[sAb * b + (long)m * lda + k];
        int seg = k >> 7, k2 = k & 127;
        float av = ai[(long)m * 128 + k2];
        float qq = q2d ? qv[(long)(m / Lq) * 128 + k2] : qv[(long)m * 128 + k2];
        return (seg == 0) ? av : (seg == 1) ? qq : (seg == 2) ? (av - qq) : (av * qq);
    };
    auto fetchB = [&](int k, int n) -> float {
        if (k >= K || n >= N) return 0.f;
        return BTRANS ? Bm[sBb * b + (long)n * ldb + k]
                      : Bm[sBb * b + (long)k * ldb + n];
    };

    v8f acc = {};
    const int KT = (K + 31) >> 5;
    for (int kb = 0; kb < KT; ++kb) {
        const int k0 = kb << 5;
        if (AMODE == 0 && kb + 1 < KT) {            // speculative global_prefetch_b8
            int mp = M0 + lane;
            if (mp < M) __builtin_prefetch(&A[sAb * b + (long)mp * lda + k0 + 32], 0, 1);
        }
        #pragma unroll
        for (int i = 0; i < 4; ++i) {
            int e = tid + (i << 7);                 // 0..511 pair slots
            {   // A: consecutive tids -> consecutive k (coalesced)
                int mm = e >> 4, kp = e & 15;
                int k = k0 + (kp << 1);
                As32[mm][kp] = pk_bf16(fetchA(M0 + mm, k), fetchA(M0 + mm, k + 1));
            }
            {   // B: consecutive tids -> consecutive n (coalesced)
                int nn = e & 31, kp = e >> 5;
                int k = k0 + (kp << 1);
                Bs32[nn][kp] = pk_bf16(fetchB(k, N0 + nn), fetchB(k + 1, N0 + nn));
            }
        }
        __syncthreads();
        v16bf afrag = ld_frag_lds(&As32[wm + l15][0], g * 4, 8 + g * 4);
        v16bf bfrag = ld_frag_lds(&Bs32[wn + l15][0], g * 4, 8 + g * 4);
        acc = __builtin_amdgcn_wmma_f32_16x16x32_bf16(false, afrag, false, bfrag,
                                                      (short)0, acc, false, false);
        __syncthreads();
    }
    #pragma unroll
    for (int v = 0; v < 8; ++v) {
        int mrow = M0 + wm + (g << 3) + v;
        int ncol = N0 + wn + l15;
        if (mrow < M && ncol < N) {
            float val = acc[v];
            if (EPI == 1) val += bias[ncol];
            else if (EPI == 2) {
                val += Cadd[sDb * b + (long)mrow * ldc + ncol];
                val = val > 0.f ? val : 0.01f * val;
            }
            C[sCb * b + (long)mrow * ldc + ncol] = val;
        }
    }
}

// ---------------- embedding / setup ----------------
__global__ void k_embed(const int* __restrict__ hp, const int* __restrict__ hy,
                        const int* __restrict__ hl, const int* __restrict__ cp,
                        const float* __restrict__ ie, const float* __restrict__ ae,
                        float* __restrict__ X, float* __restrict__ tgt, float* __restrict__ mask)
{
    int l = blockIdx.x, b = blockIdx.y, d = threadIdx.x;
    long idx = (long)b * L_ + l;
    int pro = hp[idx], y = hy[idx];
    float v = (d < 64) ? ie[(long)pro * 64 + d] : ae[(long)y * 64 + (d - 64)];
    X[idx * 128 + d] = v;
    if (d == 0) mask[idx] = (l < hl[b]) ? 1.f : 0.f;
    if (l == 0) {
        int c = cp[b];
        tgt[(long)b * 128 + d] = (d < 64) ? ie[(long)c * 64 + d] : 0.f;
    }
}

__global__ void k_xf(const float* __restrict__ X, const float* __restrict__ wt,
                     float* __restrict__ Xf)
{
    __shared__ float red[128];
    int l = blockIdx.x, b = blockIdx.y, d = threadIdx.x;
    long base = ((long)b * L_ + l) * 128;
    float v = X[base + d] * wt[d];
    red[d] = v * v; __syncthreads();
    for (int s = 64; s > 0; s >>= 1) { if (d < s) red[d] += red[d + s]; __syncthreads(); }
    float nrm = fmaxf(sqrtf(red[0]), 1e-12f);
    Xf[base + d] = v / nrm;
}

// ---------------- per-row min-max scale of S, then mask ----------------
__global__ void k_rowscale(float* __restrict__ S, const float* __restrict__ mask)
{
    __shared__ float rmn[256], rmx[256];
    int i = blockIdx.x, b = blockIdx.y, t = threadIdx.x;
    long base = ((long)b * L_ + i) * L_;
    float v = (t < L_) ? S[base + t] : 0.f;
    rmn[t] = (t < L_) ? v : 1e30f;
    rmx[t] = (t < L_) ? v : -1e30f;
    __syncthreads();
    for (int s = 128; s > 0; s >>= 1) {
        if (t < s) { rmn[t] = fminf(rmn[t], rmn[t + s]); rmx[t] = fmaxf(rmx[t], rmx[t + s]); }
        __syncthreads();
    }
    float mn = rmn[0], mx = rmx[0];
    if (t < L_) {
        float den = mx - mn;
        float sc = (den > 0.f) ? (v - mn) / den : 0.f;
        S[base + t] = sc * mask[(long)b * L_ + i] * mask[(long)b * L_ + t];
    }
}

// ---------------- radix-select k-th largest per batch (nonneg floats) ---
__global__ void k_select(const float* __restrict__ S, float* __restrict__ thr)
{
    __shared__ int hist[256];
    __shared__ int redI[256];
    __shared__ int selbin, remS;
    int b = blockIdx.x, t = threadIdx.x;
    const float* s = S + (long)b * ((long)L_ * L_);
    int cnt = 0;
    for (int i = t; i < L_ * L_; i += 256) cnt += (s[i] != 0.f) ? 1 : 0;
    redI[t] = cnt; __syncthreads();
    for (int st = 128; st > 0; st >>= 1) { if (t < st) redI[t] += redI[t + st]; __syncthreads(); }
    int k = (redI[0] + 1) >> 1;               // ceil(num_edges*0.5), desc rank (0-indexed)
    int rem = k;
    unsigned prefix = 0;
    for (int shift = 24; shift >= 0; shift -= 8) {
        hist[t] = 0; __syncthreads();
        unsigned hm = (shift == 24) ? 0u : (0xFFFFFFFFu << (shift + 8));
        for (int i = t; i < L_ * L_; i += 256) {
            unsigned key = __builtin_bit_cast(unsigned, s[i]);
            if ((key & hm) == prefix) atomicAdd(&hist[(key >> shift) & 255], 1);
        }
        __syncthreads();
        if (t == 0) {
            int r = rem, bsel = 0;
            for (int bin = 255; bin >= 0; --bin) {
                if (hist[bin] > r) { bsel = bin; break; }
                r -= hist[bin];
            }
            selbin = bsel; remS = r;
        }
        __syncthreads();
        prefix |= ((unsigned)selbin) << shift;
        rem = remS;
        __syncthreads();
    }
    if (t == 0) thr[b] = __builtin_bit_cast(float, prefix);
}

// ---------------- adjacency: bool, degrees, sym-normalize -------------
__global__ void k_adjbool(float* __restrict__ S, const float* __restrict__ thr)
{
    long idx = (long)blockIdx.x * blockDim.x + threadIdx.x;
    long total = (long)B_ * L_ * L_;
    if (idx >= total) return;
    int b = (int)(idx / ((long)L_ * L_));
    int r = (int)((idx / L_) % L_);
    int c = (int)(idx % L_);
    S[idx] = (r == c) ? 1.f : ((S[idx] > thr[b]) ? 1.f : 0.f);
}

__global__ void k_deg(const float* __restrict__ A, float* __restrict__ dis)
{
    __shared__ float red[256];
    int i = blockIdx.x, b = blockIdx.y, t = threadIdx.x;
    long base = ((long)b * L_ + i) * L_;
    red[t] = (t < L_) ? A[base + t] : 0.f; __syncthreads();
    for (int s = 128; s > 0; s >>= 1) { if (t < s) red[t] += red[t + s]; __syncthreads(); }
    if (t == 0) dis[(long)b * L_ + i] = rsqrtf(red[0]);
}

__global__ void k_anscale(float* __restrict__ A, const float* __restrict__ dis)
{
    long idx = (long)blockIdx.x * blockDim.x + threadIdx.x;
    long total = (long)B_ * L_ * L_;
    if (idx >= total) return;
    int b = (int)(idx / ((long)L_ * L_));
    int r = (int)((idx / L_) % L_);
    int c = (int)(idx % L_);
    A[idx] *= dis[(long)b * L_ + r] * dis[(long)b * L_ + c];
}

// ---------------- edge softmax E ----------------
__global__ void k_E(const float* __restrict__ An, const float* __restrict__ f1,
                    const float* __restrict__ f2, float* __restrict__ E)
{
    __shared__ float red[256];
    int i = blockIdx.x, b = blockIdx.y, t = threadIdx.x;
    long base = ((long)b * L_ + i) * L_;
    bool ab = (t < L_) && (An[base + t] > 0.f);
    float val = -1e30f;
    if (ab) {
        float e = f1[(long)b * L_ + t] + f2[(long)b * L_ + i];
        val = e > 0.f ? e : 0.01f * e;            // leaky_relu(0.01)
    }
    red[t] = val; __syncthreads();
    for (int s = 128; s > 0; s >>= 1) { if (t < s) red[t] = fmaxf(red[t], red[t + s]); __syncthreads(); }
    float mx = red[0]; __syncthreads();
    float ex = ab ? __expf(val - mx) : 0.f;
    red[t] = ex; __syncthreads();
    for (int s = 128; s > 0; s >>= 1) { if (t < s) red[t] += red[t + s]; __syncthreads(); }
    float sm = red[0];
    if (t < L_) E[base + t] = ab ? ex / sm : 0.f;
}

// ---------------- masked softmax over L (optional o2 add, optional *mask) --
__global__ void k_softmax(const float* __restrict__ o, const float* __restrict__ o2,
                          const float* __restrict__ mask, float* __restrict__ f, int mulmask)
{
    __shared__ float red[256];
    int b = blockIdx.x, t = threadIdx.x;
    long base = (long)b * L_;
    bool valid = (t < L_) && (mask[base + t] > 0.f);
    float val = -1e30f;
    if (valid) val = o[base + t] + (o2 ? o2[base + t] : 0.f);
    red[t] = val; __syncthreads();
    for (int s = 128; s > 0; s >>= 1) { if (t < s) red[t] = fmaxf(red[t], red[t + s]); __syncthreads(); }
    float mx = red[0]; __syncthreads();
    float ex = valid ? __expf(val - mx) : 0.f;
    red[t] = ex; __syncthreads();
    for (int s = 128; s > 0; s >>= 1) { if (t < s) red[t] += red[t + s]; __syncthreads(); }
    float sm = red[0];
    float inv = sm > 0.f ? 1.f / sm : 0.f;
    if (t < L_) {
        float r = ex * inv;
        if (mulmask) r *= mask[base + t];
        f[base + t] = r;
    }
}

// ---------------- BatchNorm (training stats) ----------------
__global__ void k_zero(float* __restrict__ p, long n)
{
    for (long i = threadIdx.x; i < n; i += blockDim.x) p[i] = 0.f;
}
__global__ void k_bnacc(const float* __restrict__ z, long rows, int C, float* __restrict__ st)
{
    long idx = (long)blockIdx.x * blockDim.x + threadIdx.x;
    if (idx >= rows * C) return;
    int c = (int)(idx % C);
    float v = z[idx];
    atomicAdd(&st[c], v);
    atomicAdd(&st[C + c], v * v);
}
__global__ void k_bnapply(float* __restrict__ z, long rows, int C,
                          const float* __restrict__ st,
                          const float* __restrict__ gm, const float* __restrict__ be)
{
    long idx = (long)blockIdx.x * blockDim.x + threadIdx.x;
    if (idx >= rows * C) return;
    int c = (int)(idx % C);
    float m = st[c] / (float)rows;
    float v = st[C + c] / (float)rows - m * m;
    float val = (z[idx] - m) * rsqrtf(v + BN_EPS) * gm[c] + be[c];
    z[idx] = fmaxf(val, 0.f);                    // fused ReLU
}

// ---------------- small row-dot (final FCN layer, width<=80) ----------
__global__ void k_rowdot(const float* __restrict__ z, int C,
                         const float* __restrict__ wv, const float* __restrict__ bv,
                         float* __restrict__ out, long rows)
{
    long r = (long)blockIdx.x * blockDim.x + threadIdx.x;
    if (r >= rows) return;
    float acc = 0.f;
    for (int c = 0; c < C; ++c) acc += z[r * C + c] * wv[c];
    out[r] = acc + bv[0];
}

// ---------------- pooling: bitonic sort 200 scores, threshold, mask2 ---
__global__ void k_pool(const float* __restrict__ cs, const float* __restrict__ mask,
                       float* __restrict__ mask2, float* __restrict__ csm)
{
    __shared__ float arr[256];
    __shared__ float red[256];
    int b = blockIdx.x, t = threadIdx.x;
    long base = (long)b * L_;
    float c = (t < L_) ? cs[base + t] : -1e30f;
    red[t] = (t < L_) ? mask[base + t] : 0.f; __syncthreads();
    for (int s = 128; s > 0; s >>= 1) { if (t < s) red[t] += red[t + s]; __syncthreads(); }
    int tk = (int)fminf(red[0], 30.f);
    __syncthreads();
    arr[t] = c; __syncthreads();
    for (int k = 2; k <= 256; k <<= 1)
        for (int j = k >> 1; j > 0; j >>= 1) {
            int ixj = t ^ j;
            if (ixj > t) {
                float a = arr[t], bb = arr[ixj];
                bool up = ((t & k) == 0);
                if (up ? (a < bb) : (a > bb)) { arr[t] = bb; arr[ixj] = a; }
            }
            __syncthreads();
        }
    float thrv = arr[tk];
    if (t < L_) {
        float m2 = (c > thrv) ? 1.f : 0.f;
        mask2[base + t] = m2;
        csm[base + t]   = c * m2;
    }
}

__global__ void k_readout(const float* __restrict__ Xc, const float* __restrict__ csm,
                          float* __restrict__ gr)
{
    int b = blockIdx.x, d = threadIdx.x;
    float acc = 0.f;
    for (int l = 0; l < L_; ++l) acc += Xc[((long)b * L_ + l) * 128 + d] * csm[(long)b * L_ + l];
    gr[(long)b * 128 + d] = acc;
}

// ---------------- stable partition (kept nodes to front) --------------
__global__ void k_partition(const float* __restrict__ Xc, const float* __restrict__ mask2,
                            float* __restrict__ Xs, float* __restrict__ mask_s,
                            int* __restrict__ redlen)
{
    __shared__ int pre[256];
    __shared__ int dest[256];
    int b = blockIdx.x, t = threadIdx.x;
    long base = (long)b * L_;
    int m = (t < L_) ? (int)mask2[base + t] : 0;
    pre[t] = m; __syncthreads();
    for (int off = 1; off < 256; off <<= 1) {
        int v = pre[t];
        int add = (t >= off) ? pre[t - off] : 0;
        __syncthreads();
        pre[t] = v + add;
        __syncthreads();
    }
    int kept = pre[255];
    if (t < L_) dest[t] = m ? (pre[t] - 1) : (kept + t - pre[t]);
    if (t == 0) redlen[b] = kept;
    __syncthreads();
    for (int i = t; i < L_ * 128; i += 256) {
        int l = i >> 7, d = i & 127;
        Xs[(base + dest[l]) * 128 + d] = Xc[(base + l) * 128 + d];
    }
    if (t < L_) mask_s[base + dest[t]] = (float)m;
}

// ---------------- pack Wh^T into bf16 pairs: WhT[n][kpair] ------------
__global__ void k_packWh(const float* __restrict__ Wh, unsigned* __restrict__ WhT)
{
    int idx = blockIdx.x * blockDim.x + threadIdx.x;    // 384*64
    if (idx >= 384 * 64) return;
    int n = idx >> 6, kp = idx & 63;
    WhT[idx] = pk_bf16(Wh[(long)(2 * kp) * 384 + n], Wh[(long)(2 * kp + 1) * 384 + n]);
}

// ---------------- persistent AUGRU: 16 batch rows / WG, WMMA h@Wh ------
__global__ void __launch_bounds__(256)
k_augru(const float* __restrict__ xp, const float* __restrict__ alph,
        const unsigned* __restrict__ WhT, const float* __restrict__ bh,
        const int* __restrict__ redlen, float* __restrict__ fin)
{
    __shared__ float hS[16][129];                     // fp32 hidden state
    __shared__ __align__(16) unsigned hB[16][68];     // bf16-pair mirror of h
    __shared__ float h3S[16][385];
    int b0 = blockIdx.x * 16;
    int tid = threadIdx.x, lane = tid & 31, w = tid >> 5;   // 8 wave32
    int g = lane >> 4, l15 = lane & 15;
    for (int i = tid; i < 16 * 128; i += 256) hS[i >> 7][i & 127] = 0.f;
    for (int i = tid; i < 16 * 68;  i += 256) hB[i / 68][i % 68] = 0u;
    __syncthreads();
    for (int t = 0; t < L_; ++t) {
        // h3 = h @ Wh   (16 x 384, 24 tiles, 3 per wave)
        for (int it = 0; it < 3; ++it) {
            int n0 = ((w * 3 + it) << 4);
            v8f acc = {};
            #pragma unroll
            for (int kb = 0; kb < 4; ++kb) {
                int kp0 = kb << 4;                    // k0/2
                v16bf afrag = ld_frag_lds(&hB[l15][0], kp0 + g * 4, kp0 + 8 + g * 4);
                v16bf bfrag = ld_frag_glb(WhT + (long)(n0 + l15) * 64,
                                          kp0 + g * 4, kp0 + 8 + g * 4);
                acc = __builtin_amdgcn_wmma_f32_16x16x32_bf16(false, afrag, false, bfrag,
                                                              (short)0, acc, false, false);
            }
            #pragma unroll
            for (int v = 0; v < 8; ++v) {
                int m = (g << 3) + v;
                int n = n0 + l15;
                h3S[m][n] = acc[v] + bh[n];
            }
        }
        __syncthreads();
        // fused gates, element pairs (also refresh bf16 mirror)
        for (int i = tid; i < 16 * 64; i += 256) {
            int m = i >> 6, dp = i & 63, d = dp << 1;
            int bb = b0 + m;
            long basex = ((long)bb * L_ + t) * 384;
            float a = alph[(long)bb * L_ + t];
            bool last = (t == redlen[bb] - 1);
            float hn01[2];
            #pragma unroll
            for (int q = 0; q < 2; ++q) {
                int dd = d + q;
                float xr = xp[basex + dd], xu = xp[basex + 128 + dd], xn = xp[basex + 256 + dd];
                float hr = h3S[m][dd], hu = h3S[m][128 + dd], hn = h3S[m][256 + dd];
                float r  = 1.f / (1.f + __expf(-(xr + hr)));
                float u  = a / (1.f + __expf(-(xu + hu)));
                float nn = tanhf(xn + r * hn);
                float h  = hS[m][dd];
                float hnew = (1.f - u) * h + u * nn;
                hS[m][dd] = hnew;
                hn01[q] = hnew;
                if (last) fin[(long)bb * 128 + dd] = hnew;
            }
            hB[m][dp] = pk_bf16(hn01[0], hn01[1]);
        }
        __syncthreads();
    }
}

__global__ void k_mo(const float* __restrict__ fin, const float* __restrict__ gr,
                     const float* __restrict__ tgt, float* __restrict__ mo)
{
    int b = blockIdx.x, d = threadIdx.x;
    float f = fin[(long)b * 128 + d], g = gr[(long)b * 128 + d], tg = tgt[(long)b * 128 + d];
    long o = (long)b * 512;
    mo[o + d] = f; mo[o + 128 + d] = g; mo[o + 256 + d] = tg; mo[o + 384 + d] = g * tg;
}

// =====================================================================
extern "C" void kernel_launch(void* const* d_in, const int* in_sizes, int n_in,
                              void* d_out, int out_size, void* d_ws, size_t ws_size,
                              hipStream_t stream)
{
    const int*   his_pro  = (const int*)  d_in[0];
    const int*   his_y    = (const int*)  d_in[1];
    const int*   his_len  = (const int*)  d_in[2];
    const int*   cur_pro  = (const int*)  d_in[3];
    const float* item_emb = (const float*)d_in[5];
    const float* ans_emb  = (const float*)d_in[6];
    const float* wt       = (const float*)d_in[7];
    const float* att_cl   = (const float*)d_in[8];
    const float* att_q    = (const float*)d_in[9];
    const float* att_rnn  = (const float*)d_in[10];
    const float* aggre_W  = (const float*)d_in[11];
    const float* gru_Wx   = (const float*)d_in[12];
    const float* gru_Wh   = (const float*)d_in[13];
    const float* gru_bx   = (const float*)d_in[14];
    const float* gru_bh   = (const float*)d_in[15];
    const float* a_w0 = (const float*)d_in[16]; const float* a_b0 = (const float*)d_in[17];
    const float* a_g0 = (const float*)d_in[18]; const float* a_be0= (const float*)d_in[19];
    const float* a_w1 = (const float*)d_in[20]; const float* a_b1 = (const float*)d_in[21];
    const float* a_g1 = (const float*)d_in[22]; const float* a_be1= (const float*)d_in[23];
    const float* a_wo = (const float*)d_in[24]; const float* a_bo = (const float*)d_in[25];
    const float* t_w0 = (const float*)d_in[26]; const float* t_b0 = (const float*)d_in[27];
    const float* t_g0 = (const float*)d_in[28]; const float* t_be0= (const float*)d_in[29];
    const float* t_w1 = (const float*)d_in[30]; const float* t_b1 = (const float*)d_in[31];
    const float* t_g1 = (const float*)d_in[32]; const float* t_be1= (const float*)d_in[33];
    const float* t_wo = (const float*)d_in[34]; const float* t_bo = (const float*)d_in[35];

    const long BL = (long)B_ * L_;      // 51200
    const long LL = (long)L_ * L_;      // 40000

    // ---------------- workspace bump allocator ----------------
    float* base = (float*)d_ws;
    long off = 0;
    auto alloc = [&](long n) { float* p = base + off; off += (n + 63) & ~63L; return p; };
    float* X    = alloc(BL * 128);
    float* Xc   = alloc(BL * 128);
    float* bufA = alloc(BL * 128);   // Xf / T / ai / EX / Xs
    float* bufB = alloc(BL * 128);   // Xq / ai (att5)
    float* An   = alloc((long)B_ * LL);   // S -> A_bool -> An (in place)
    float* E    = alloc((long)B_ * LL);
    float* xp   = alloc(BL * 384);
    float* z0   = alloc(BL * 80);
    float* z1   = alloc(BL * 40);
    float* tgt  = alloc((long)B_ * 128);
    float* maskb= alloc(BL);
    float* f1   = alloc(BL);
    float* f2   = alloc(BL);
    float* ob   = alloc(BL);
    float* cs   = alloc(BL);
    float* alph = alloc(BL);
    float* mask2= alloc(BL);
    float* masks= alloc(BL);
    float* csm  = alloc(BL);
    float* dis  = alloc(BL);
    float* thr  = alloc(B_);
    float* fin  = alloc((long)B_ * 128);
    float* gr   = alloc((long)B_ * 128);
    float* mo   = alloc((long)B_ * 512);
    float* hz0  = alloc((long)B_ * 200);
    float* hz1  = alloc((long)B_ * 80);
    float* stats= alloc(512);
    unsigned* WhT = (unsigned*)alloc(384 * 64);
    int*   redl = (int*)alloc(B_);
    (void)ws_size; (void)in_sizes; (void)n_in; (void)out_size;

    // epi: 0 none, 1 bias, 2 residual+leaky; amode 1 => fused attention input
    auto gemm = [&](const float* A, const float* Bm, float* C,
                    int M, int N, int K, int lda, int ldb, int ldc,
                    long sA, long sB, long sC, int bat,
                    int amode, const float* ai, const float* qv, int q2d,
                    int btrans, int epi, const float* bias, const float* Cadd, long sD) {
        dim3 grd((M + 31) / 32, (N + 31) / 32, bat);
        dim3 blk(128);
        if (amode == 1)
            k_gemm_t<1,0,1><<<grd, blk, 0, stream>>>(A, Bm, C, M, N, K, lda, ldb, ldc,
                                                     sA, sB, sC, ai, qv, q2d, L_, bias, Cadd, sD);
        else if (btrans)
            k_gemm_t<0,1,0><<<grd, blk, 0, stream>>>(A, Bm, C, M, N, K, lda, ldb, ldc,
                                                     sA, sB, sC, ai, qv, q2d, L_, bias, Cadd, sD);
        else if (epi == 0)
            k_gemm_t<0,0,0><<<grd, blk, 0, stream>>>(A, Bm, C, M, N, K, lda, ldb, ldc,
                                                     sA, sB, sC, ai, qv, q2d, L_, bias, Cadd, sD);
        else if (epi == 1)
            k_gemm_t<0,0,1><<<grd, blk, 0, stream>>>(A, Bm, C, M, N, K, lda, ldb, ldc,
                                                     sA, sB, sC, ai, qv, q2d, L_, bias, Cadd, sD);
        else
            k_gemm_t<0,0,2><<<grd, blk, 0, stream>>>(A, Bm, C, M, N, K, lda, ldb, ldc,
                                                     sA, sB, sC, ai, qv, q2d, L_, bias, Cadd, sD);
    };
    auto bn = [&](float* z, long rows, int C, const float* gm, const float* be) {
        k_zero<<<1, 256, 0, stream>>>(stats, (long)(2 * C));
        long tot = rows * C;
        int blks = (int)((tot + 255) / 256);
        k_bnacc  <<<blks, 256, 0, stream>>>(z, rows, C, stats);
        k_bnapply<<<blks, 256, 0, stream>>>(z, rows, C, stats, gm, be);
    };
    auto attention = [&](const float* kv, const float* mat, const float* q, int q2d,
                         float* aiBuf, const float* mskp, float* fout) {
        // ai = kv @ mat
        gemm(kv, mat, aiBuf, (int)BL, 128, 128, 128, 128, 128, 0, 0, 0, 1,
             0, nullptr, nullptr, 0, 0, 0, nullptr, nullptr, 0);
        // z0 = [ai,q,ai-q,ai*q] @ w0 + b0   (A built on the fly)
        gemm(nullptr, a_w0, z0, (int)BL, 80, 512, 0, 80, 80, 0, 0, 0, 1,
             1, aiBuf, q, q2d, 0, 1, a_b0, nullptr, 0);
        bn(z0, BL, 80, a_g0, a_be0);
        gemm(z0, a_w1, z1, (int)BL, 40, 80, 80, 40, 40, 0, 0, 0, 1,
             0, nullptr, nullptr, 0, 0, 1, a_b1, nullptr, 0);
        bn(z1, BL, 40, a_g1, a_be1);
        k_rowdot<<<(int)((BL + 255) / 256), 256, 0, stream>>>(z1, 40, a_wo, a_bo, ob, BL);
        k_softmax<<<B_, 256, 0, stream>>>(ob, nullptr, mskp, fout, 0);
    };

    // ============ pipeline ============
    k_embed<<<dim3(L_, B_), 128, 0, stream>>>(his_pro, his_y, his_len, cur_pro,
                                              item_emb, ans_emb, X, tgt, maskb);
    k_xf<<<dim3(L_, B_), 128, 0, stream>>>(X, wt, bufA);
    // S = Xf @ Xf^T  (batched NT)
    gemm(bufA, bufA, An, L_, L_, 128, 128, 128, L_, (long)L_ * 128, (long)L_ * 128, LL, B_,
         0, nullptr, nullptr, 0, 1, 0, nullptr, nullptr, 0);
    k_rowscale<<<dim3(L_, B_), 256, 0, stream>>>(An, maskb);
    k_select<<<B_, 256, 0, stream>>>(An, thr);
    {
        int blks = (int)(((long)B_ * LL + 255) / 256);
        k_adjbool<<<blks, 256, 0, stream>>>(An, thr);
        k_deg<<<dim3(L_, B_), 256, 0, stream>>>(An, dis);
        k_anscale<<<blks, 256, 0, stream>>>(An, dis);
    }
    // Xq = An @ (An @ X)
    gemm(An, X, bufA, L_, 128, L_, L_, 128, 128, LL, (long)L_ * 128, (long)L_ * 128, B_,
         0, nullptr, nullptr, 0, 0, 0, nullptr, nullptr, 0);
    gemm(An, bufA, bufB, L_, 128, L_, L_, 128, 128, LL, (long)L_ * 128, (long)L_ * 128, B_,
         0, nullptr, nullptr, 0, 0, 0, nullptr, nullptr, 0);
    attention(X, att_cl, bufB, 0, bufA, maskb, f1);       // query = Xq (3D)
    attention(X, att_q,  tgt,  1, bufA, maskb, f2);       // query = tgt (2D)
    k_E<<<dim3(L_, B_), 256, 0, stream>>>(An, f1, f2, E);
    // Xc = leaky(E@X @ aggre_W + X)
    gemm(E, X, bufA, L_, 128, L_, L_, 128, 128, LL, (long)L_ * 128, (long)L_ * 128, B_,
         0, nullptr, nullptr, 0, 0, 0, nullptr, nullptr, 0);
    gemm(bufA, aggre_W, Xc, (int)BL, 128, 128, 128, 128, 128, 0, 0, 0, 1,
         0, nullptr, nullptr, 0, 0, 2, nullptr, X, 0);
    // Xq2 = An @ (An @ Xc)
    gemm(An, Xc, bufA, L_, 128, L_, L_, 128, 128, LL, (long)L_ * 128, (long)L_ * 128, B_,
         0, nullptr, nullptr, 0, 0, 0, nullptr, nullptr, 0);
    gemm(An, bufA, bufB, L_, 128, L_, L_, 128, 128, LL, (long)L_ * 128, (long)L_ * 128, B_,
         0, nullptr, nullptr, 0, 0, 0, nullptr, nullptr, 0);
    attention(Xc, att_cl, bufB, 0, bufA, maskb, f1);
    attention(Xc, att_q,  tgt,  1, bufA, maskb, f2);
    // cluster scores, pooling, readout
    k_softmax<<<B_, 256, 0, stream>>>(f1, f2, maskb, cs, 1);
    k_pool<<<B_, 256, 0, stream>>>(cs, maskb, mask2, csm);
    k_readout<<<B_, 128, 0, stream>>>(Xc, csm, gr);
    // stable partition -> Xs, mask_s, red_len
    k_partition<<<B_, 256, 0, stream>>>(Xc, mask2, bufA, masks, redl);
    attention(bufA, att_rnn, tgt, 1, bufB, masks, alph);
    // xp = Xs @ Wx + bx, then persistent AUGRU over bf16-packed Wh^T
    gemm(bufA, gru_Wx, xp, (int)BL, 384, 128, 128, 384, 384, 0, 0, 0, 1,
         0, nullptr, nullptr, 0, 0, 1, gru_bx, nullptr, 0);
    k_packWh<<<(384 * 64 + 255) / 256, 256, 0, stream>>>(gru_Wh, WhT);
    k_augru<<<B_ / 16, 256, 0, stream>>>(xp, alph, WhT, gru_bh, redl, fin);
    // prediction head
    k_mo<<<B_, 128, 0, stream>>>(fin, gr, tgt, mo);
    gemm(mo, t_w0, hz0, B_, 200, 512, 512, 200, 200, 0, 0, 0, 1,
         0, nullptr, nullptr, 0, 0, 1, t_b0, nullptr, 0);
    bn(hz0, B_, 200, t_g0, t_be0);
    gemm(hz0, t_w1, hz1, B_, 80, 200, 200, 80, 80, 0, 0, 0, 1,
         0, nullptr, nullptr, 0, 0, 1, t_b1, nullptr, 0);
    bn(hz1, B_, 80, t_g1, t_be1);
    k_rowdot<<<1, 256, 0, stream>>>(hz1, 80, t_wo, t_bo, (float*)d_out, (long)B_);
}
